// Projector_36163624632784
// MI455X (gfx1250) — matrix operand order repaired and checked by reading, not compiled, not run on hardware
//
#include <hip/hip_runtime.h>
#include <hip/hip_bf16.h>

// MI455X (gfx1250) fused dilated-window attention.
// One workgroup (256 thr = 8 waves) per (b, g, r) unit; all intermediates in LDS.
// bf16 WMMA (v_wmma_f32_16x16x32_bf16) with f32 accumulation throughout.

typedef __attribute__((ext_vector_type(16))) __bf16 bf16x16;
typedef __attribute__((ext_vector_type(8)))  float  f32x8;

union Frag {
    bf16x16        v;
    uint4          q[2];
    unsigned short us[16];
};

__device__ __forceinline__ unsigned short f2bf(float x) {
    unsigned int u = __float_as_uint(x);
    u += 0x7FFFu + ((u >> 16) & 1u);      // round-to-nearest-even bf16
    return (unsigned short)(u >> 16);
}

__device__ __forceinline__ unsigned int f2bf2(float lo, float hi) {
    return (unsigned int)f2bf(lo) | ((unsigned int)f2bf(hi) << 16);
}

__device__ __forceinline__ f32x8 wmma_bf16(const Frag& a, const Frag& b, f32x8 c) {
    return __builtin_amdgcn_wmma_f32_16x16x32_bf16(false, a.v, false, b.v,
                                                   (short)0, c, false, false);
}

// LDS map (ushort units). 32768 us = 64 KB total.
#define SM_XQO 0        // [64][256] bf16: emb tile -> Q (in place) -> attn out (in place)
#define SM_WP  16384    // 32 KB shared: W tile [256][32] during GEMMs; P [4][64][64] during attention
#define LDS_US 32768

// out[M][n] = sum_k A[M][k] * W[n][k] + bias[n], A = LDS [64][256] bf16, M<=64, n=256.
// Wave w: M-tile = w&3, 8 N-tiles starting at 8*(w>>2). K staged in 32-wide bf16 tiles.
template<bool TO_GLOBAL>
__device__ __forceinline__ void gemm_AxWT(unsigned short* smem,
                                          const float* __restrict__ W,
                                          const float* __restrict__ bias,
                                          float* __restrict__ out_g, int tid) {
    unsigned short* A  = smem + SM_XQO;
    unsigned short* Wt = smem + SM_WP;           // Bt[n][kk] = W[n][k0+kk]
    const int lane = tid & 31, wave = tid >> 5;
    const int half = lane >> 4, l15 = lane & 15;
    const int mt = wave & 3;
    const int jbase = (wave >> 2) * 8;

    f32x8 acc[8] = {};
    for (int k0 = 0; k0 < 256; k0 += 32) {
        __syncthreads();
        // prefetch next k-tile of W toward the WGP while this tile computes
        if (k0 < 224) __builtin_prefetch(&W[tid * 256 + k0 + 32]);
        // stage 256x32 fp32 -> bf16, float4 reads + packed uint2 LDS stores
        for (int e4 = tid; e4 < 2048; e4 += 256) {
            const int n  = e4 >> 3;
            const int kk = (e4 & 7) << 2;
            const float4 w = *(const float4*)&W[n * 256 + k0 + kk];
            uint2 pk;
            pk.x = f2bf2(w.x, w.y);
            pk.y = f2bf2(w.z, w.w);
            *(uint2*)&Wt[(n << 5) + kk] = pk;
        }
        __syncthreads();
        Frag a;                                  // A frag: rows 16*mt + l15
        const char* Ab = (const char*)A + (((mt << 4) + l15) << 9) + (k0 << 1) + (half << 4);
        a.q[0] = *(const uint4*)(Ab);
        a.q[1] = *(const uint4*)(Ab + 32);
#pragma unroll
        for (int g4 = 0; g4 < 2; ++g4) {         // batch B-frag loads, then 4 WMMAs back-to-back
            Frag bfr[4];
#pragma unroll
            for (int t = 0; t < 4; ++t) {
                const int jt = jbase + g4 * 4 + t;
                const char* Bb = (const char*)Wt + (((jt << 4) + l15) << 6) + (half << 4);
                bfr[t].q[0] = *(const uint4*)(Bb);
                bfr[t].q[1] = *(const uint4*)(Bb + 32);
            }
#pragma unroll
            for (int t = 0; t < 4; ++t)
                acc[g4 * 4 + t] = wmma_bf16(a, bfr[t], acc[g4 * 4 + t]);
        }
    }
    __syncthreads();   // all waves done reading A/Wt before A is overwritten in place
#pragma unroll
    for (int t = 0; t < 8; ++t) {
        const int ncol = ((jbase + t) << 4) + l15;
        const float bv = bias[ncol];
#pragma unroll
        for (int rr = 0; rr < 8; ++rr) {
            const int M = (mt << 4) + rr + (half << 3);
            const float val = acc[t][rr] + bv;
            if (TO_GLOBAL) {
                if (M < 49) out_g[(size_t)M * 1280 + ncol] = val;   // token stride = R*D
            } else {
                A[(M << 8) + ncol] = f2bf(val);
            }
        }
    }
}

__global__ void __launch_bounds__(256)
dilated_attn_fused(const float* __restrict__ emb, const float* __restrict__ Wq,
                   const float* __restrict__ bq,  const float* __restrict__ Wo,
                   const float* __restrict__ bo,  float* __restrict__ out) {
    __shared__ alignas(16) unsigned short smem[LDS_US];
    const int tid = threadIdx.x;
    const int bx  = blockIdx.x;
    const int r = bx % 5;
    const int g = (bx / 5) % 20;
    const int b = bx / 100;
    const size_t tokBase = (size_t)b * 4900 + g * 245 + r;

    // ---- stage emb rows (49 tokens, zero-padded to 64) as bf16, float4-vectorized ----
    for (int e4 = tid; e4 < 4096; e4 += 256) {
        const int row = e4 >> 6;
        const int k4  = (e4 & 63) << 2;
        float4 v = make_float4(0.f, 0.f, 0.f, 0.f);
        if (row < 49) v = *(const float4*)&emb[(tokBase + (size_t)row * 5) * 256 + k4];
        uint2 pk;
        pk.x = f2bf2(v.x, v.y);
        pk.y = f2bf2(v.z, v.w);
        *(uint2*)&smem[SM_XQO + (row << 8) + k4] = pk;
    }

    // ---- Q = Xe * Wq^T + bq (overwrites Xe in place) ----
    gemm_AxWT<false>(smem, Wq, bq, nullptr, tid);
    __syncthreads();

    const int lane = tid & 31, wave = tid >> 5;
    const int half = lane >> 4, l15 = lane & 15;
    const int h = wave >> 1, p = wave & 1;           // 2 waves per head; wave p owns rows 32p..32p+31
    unsigned short* Q = smem + SM_XQO;               // [64][256] bf16
    unsigned short* P = smem + SM_WP + (h << 12);    // [64][64] bf16 per head

    // ---- S = Xh * Xh^T (Xh = Q[:, 64h:64h+64]), 4x4 tiles, K=64 ----
    f32x8 sacc[2][4] = {};
#pragma unroll
    for (int ks = 0; ks < 2; ++ks) {
        const int kk0 = ks << 5;
        Frag a[2], bfr[4];
#pragma unroll
        for (int mi = 0; mi < 2; ++mi) {
            const char* Ab = (const char*)Q + (((((2 * p + mi) << 4) + l15) << 9))
                             + (h << 7) + (kk0 << 1) + (half << 4);
            a[mi].q[0] = *(const uint4*)(Ab);
            a[mi].q[1] = *(const uint4*)(Ab + 32);
        }
#pragma unroll
        for (int jt = 0; jt < 4; ++jt) {             // Bt[n=j][k=d] = Q[j][64h+d]: contiguous
            const char* Bb = (const char*)Q + ((((jt << 4) + l15) << 9))
                             + (h << 7) + (kk0 << 1) + (half << 4);
            bfr[jt].q[0] = *(const uint4*)(Bb);
            bfr[jt].q[1] = *(const uint4*)(Bb + 32);
        }
#pragma unroll
        for (int jt = 0; jt < 4; ++jt)
#pragma unroll
            for (int mi = 0; mi < 2; ++mi)
                sacc[mi][jt] = wmma_bf16(a[mi], bfr[jt], sacc[mi][jt]);
    }

    // ---- softmax rows: scale = dh^-0.5 = 0.125, mask cols >= 49; write P as bf16 ----
#pragma unroll
    for (int mi = 0; mi < 2; ++mi) {
#pragma unroll
        for (int rr = 0; rr < 8; ++rr) {
            float sv[4], pv[4];
            float mx = -3.0e38f;
#pragma unroll
            for (int jt = 0; jt < 4; ++jt) {
                const int col = (jt << 4) + l15;
                sv[jt] = sacc[mi][jt][rr] * 0.125f;
                mx = (col < 49) ? fmaxf(mx, sv[jt]) : mx;
            }
            mx = fmaxf(mx, __shfl_xor(mx, 1, 32));
            mx = fmaxf(mx, __shfl_xor(mx, 2, 32));
            mx = fmaxf(mx, __shfl_xor(mx, 4, 32));
            mx = fmaxf(mx, __shfl_xor(mx, 8, 32));
            float sum = 0.f;
#pragma unroll
            for (int jt = 0; jt < 4; ++jt) {
                const int col = (jt << 4) + l15;
                pv[jt] = (col < 49) ? __expf(sv[jt] - mx) : 0.f;
                sum += pv[jt];
            }
            sum += __shfl_xor(sum, 1, 32);
            sum += __shfl_xor(sum, 2, 32);
            sum += __shfl_xor(sum, 4, 32);
            sum += __shfl_xor(sum, 8, 32);
            const float inv = 1.f / sum;
            const int M = ((2 * p + mi) << 4) + rr + (half << 3);
#pragma unroll
            for (int jt = 0; jt < 4; ++jt)
                P[(M << 6) + (jt << 4) + l15] = f2bf(pv[jt] * inv);
        }
    }

    // ---- O = P * Xh (each wave uses only its own P rows; Xh B-frags strided from Q) ----
    f32x8 oacc[2][4] = {};
#pragma unroll
    for (int ks = 0; ks < 2; ++ks) {
        const int kk0 = ks << 5;
        Frag a[2], bfr[4];
#pragma unroll
        for (int mi = 0; mi < 2; ++mi) {
            const char* Ab = (const char*)P + (((((2 * p + mi) << 4) + l15) << 7))
                             + (kk0 << 1) + (half << 4);
            a[mi].q[0] = *(const uint4*)(Ab);
            a[mi].q[1] = *(const uint4*)(Ab + 32);
        }
#pragma unroll
        for (int dt = 0; dt < 4; ++dt) {             // Bt[n=d][k=j] = Q[j][64h+d]: strided
            const int d = (dt << 4) + l15;
#pragma unroll
            for (int ie = 0; ie < 16; ++ie)
                bfr[dt].us[ie] = Q[((kk0 + (half << 3) + (ie & 7) + ((ie >> 3) << 4)) << 8) + (h << 6) + d];
        }
#pragma unroll
        for (int dt = 0; dt < 4; ++dt)
#pragma unroll
            for (int mi = 0; mi < 2; ++mi)
                oacc[mi][dt] = wmma_bf16(a[mi], bfr[dt], oacc[mi][dt]);
    }
    __syncthreads();   // all waves done reading Q before O overwrites it in place
#pragma unroll
    for (int mi = 0; mi < 2; ++mi) {
#pragma unroll
        for (int dt = 0; dt < 4; ++dt) {
            const int col = (h << 6) + (dt << 4) + l15;
#pragma unroll
            for (int rr = 0; rr < 8; ++rr) {
                const int M = ((2 * p + mi) << 4) + rr + (half << 3);
                Q[(M << 8) + col] = f2bf(oacc[mi][dt][rr]);
            }
        }
    }

    // ---- final: out = O * Wo^T + bo -> global (internal barrier covers O writes) ----
    gemm_AxWT<true>(smem, Wo, bo, out + tokBase * 256, tid);
}

extern "C" void kernel_launch(void* const* d_in, const int* in_sizes, int n_in,
                              void* d_out, int out_size, void* d_ws, size_t ws_size,
                              hipStream_t stream) {
    const float* emb = (const float*)d_in[0];
    const float* Wq  = (const float*)d_in[1];
    const float* bq  = (const float*)d_in[2];
    const float* Wo  = (const float*)d_in[3];
    const float* bo  = (const float*)d_in[4];
    float* out = (float*)d_out;
    (void)in_sizes; (void)n_in; (void)out_size; (void)d_ws; (void)ws_size;
    dim3 grid(3200), block(256);
    hipLaunchKernelGGL(dilated_attn_fused, grid, block, 0, stream,
                       emb, Wq, bq, Wo, bo, out);
}
